// GraphormerMultiHeadAttention_8796093022646
// MI455X (gfx1250) — matrix operand order repaired and checked by reading, not compiled, not run on hardware
//
#include <hip/hip_runtime.h>
#include <hip/hip_bf16.h>

// ---------------------------------------------------------------------------
// Graphormer MHA for gfx1250 (MI455X), wave32.
//   k0 : one-shot f32->f16 conversion (x, W*) + bias pre-sum (spatial+edge)
//   k1 : fused QKV projection, TDM (tensor_load_to_lds) weight-panel staging
//   k2 : flash-style attention (online softmax, S^T orientation)
//   k3 : output projection, same TDM staging
// ---------------------------------------------------------------------------

typedef __attribute__((ext_vector_type(16))) _Float16 v16h;
typedef __attribute__((ext_vector_type(8)))  _Float16 v8h;
typedef __attribute__((ext_vector_type(4)))  _Float16 v4h;
typedef __attribute__((ext_vector_type(8)))  float    v8f;
typedef __attribute__((ext_vector_type(4)))  float    v4f;
typedef __attribute__((ext_vector_type(4)))  unsigned int u32x4;
typedef __attribute__((ext_vector_type(4)))  int      i32x4;
typedef __attribute__((ext_vector_type(8)))  int      i32x8;

constexpr int Bdim = 16, Nseq = 512, Dmod = 1024, Hn = 16, HD = 64;
constexpr int BN = Bdim * Nseq;              // 8192 rows
constexpr float SCALE = 0.03125f;            // Dmod ** -0.5

// LDS weight panel: 64 cols x 256 k-halves, padded row stride 264 halves
// (TDM pad: interval 128 DWORDs (code 6) + amount 4 DWORDs (code 3) => +8
// halves per 256; 264*2B = 528B rows -> 16B aligned, lanes hit distinct banks)
constexpr int KPAN = 256;
constexpr int WROW = 264;

#if defined(__has_builtin)
#if __has_builtin(__builtin_amdgcn_tensor_load_to_lds)
#define HAVE_TDM 1
#endif
#endif
#ifndef HAVE_TDM
#define HAVE_TDM 0
#endif

__device__ __forceinline__ v8f wmma16(v16h a, v16h b, v8f c) {
  return __builtin_amdgcn_wmma_f32_16x16x32_f16(false, a, false, b, (short)0, c,
                                                false, false);
}

__device__ __forceinline__ v16h join8(v8h lo, v8h hi) {
  v16h r;
#pragma unroll
  for (int i = 0; i < 8; ++i) { r[i] = lo[i]; r[8 + i] = hi[i]; }
  return r;
}

// A-fragment (16x32 f16): lane M = lane%16; elems e<8: K = hh*8+e,
// e>=8: K = 16 + hh*8 + (e-8). Two contiguous 8-half chunks.
__device__ __forceinline__ v16h load_a_f16(const _Float16* row, int hh) {
  v8h c0 = *(const v8h*)(row + hh * 8);
  v8h c1 = *(const v8h*)(row + 16 + hh * 8);
  return join8(c0, c1);
}

__device__ __forceinline__ void wait_tensorcnt0() {
#if defined(__has_builtin)
#if __has_builtin(__builtin_amdgcn_s_wait_tensorcnt)
  __builtin_amdgcn_s_wait_tensorcnt(0);
#else
  asm volatile("s_wait_tensorcnt 0x0" ::: "memory");
#endif
#endif
}

// Issue one TDM DMA: 2D tile (tile_dim0=256 halves, tile_dim1=64 rows) from a
// row-major f16 matrix with row stride 1024, into LDS at lds_off with padding.
// D# layout per CDNA5 ISA.
__device__ __forceinline__ void tdm_load_panel(unsigned int lds_off,
                                               const _Float16* gsrc) {
#if HAVE_TDM
  const unsigned long long ga = (unsigned long long)(__SIZE_TYPE__)gsrc;
  u32x4 g0;
  g0[0] = 1u;                                    // count=1, user mode
  g0[1] = lds_off;                               // lds_addr (bytes)
  g0[2] = (unsigned int)(ga & 0xFFFFFFFFu);      // global_addr[31:0]
  g0[3] = (unsigned int)((ga >> 32) & 0x1FFFFFFu) | (2u << 30);  // type=2
  i32x8 g1;
  g1[0] = (1 << 16)      // data_size = 2 bytes
        | (1 << 20)      // pad_enable
        | (6 << 22)      // pad_interval: 128 DWORDs (=256 halves)
        | (3 << 25);     // pad_amount: 4 DWORDs (=8 halves)
  g1[1] = (int)(((unsigned)KPAN & 0xFFFFu) << 16); // tensor_dim0[15:0]
  g1[2] = (int)(64u << 16);                        // tensor_dim1[15:0]=64
  g1[3] = (int)(((unsigned)KPAN) << 16);           // tile_dim0 = 256
  g1[4] = 64;                                      // tile_dim1 = 64, tile_dim2=0
  g1[5] = Dmod;                                    // tensor_dim0_stride = 1024
  g1[6] = 0;
  g1[7] = 0;
  i32x4 gz = {0, 0, 0, 0};
#if __clang_major__ >= 23
  i32x8 gz8 = {0, 0, 0, 0, 0, 0, 0, 0};
  __builtin_amdgcn_tensor_load_to_lds(g0, g1, gz, gz, gz8, 0);
#else
  __builtin_amdgcn_tensor_load_to_lds(g0, g1, gz, gz, 0);
#endif
  wait_tensorcnt0();
#else
  (void)lds_off; (void)gsrc;
#endif
}

// ---------------------------------------------------------------------------
// Kernel 0a: f32 -> f16 (vectorized x4)
// ---------------------------------------------------------------------------
__global__ __launch_bounds__(256) void cvt16_kernel(const float* __restrict__ src,
                                                    _Float16* __restrict__ dst,
                                                    int n4) {
  for (int i = blockIdx.x * blockDim.x + threadIdx.x; i < n4;
       i += gridDim.x * blockDim.x) {
    v4f v = ((const v4f*)src)[i];
    v4h h;
#pragma unroll
    for (int j = 0; j < 4; ++j) h[j] = (_Float16)v[j];
    ((v4h*)dst)[i] = h;
  }
}

// Kernel 0b: bias = spatial + edge (kept f32 for softmax accuracy)
__global__ __launch_bounds__(256) void biassum_kernel(
    const float* __restrict__ sp, const float* __restrict__ ed,
    float* __restrict__ bs, int n4) {
  for (int i = blockIdx.x * blockDim.x + threadIdx.x; i < n4;
       i += gridDim.x * blockDim.x) {
    v4f a = ((const v4f*)sp)[i];
    v4f b = ((const v4f*)ed)[i];
    ((v4f*)bs)[i] = a + b;
  }
}

// ---------------------------------------------------------------------------
// Shared GEMM core: one block = 8 waves, each wave a 32x64 tile of C = A @ W^T.
// Weight panel (64 out-cols x 256 k) staged in LDS via TDM, shared by waves.
// ---------------------------------------------------------------------------
struct GemmAcc { v8f a[2][4]; };

__device__ __forceinline__ void gemm_core(const _Float16* __restrict__ A,
                                          const _Float16* __restrict__ W16,
                                          _Float16* __restrict__ wtile,
                                          int rowbase, int colbase, int wave,
                                          int lane, GemmAcc& acc) {
  const int hh = lane >> 4, l16 = lane & 15;
#pragma unroll
  for (int g = 0; g < 2; ++g)
#pragma unroll
    for (int s = 0; s < 4; ++s)
#pragma unroll
      for (int e = 0; e < 8; ++e) acc.a[g][s][e] = 0.0f;

  const _Float16* arow0 = A + (size_t)(rowbase + l16) * Dmod;
  const _Float16* arow1 = A + (size_t)(rowbase + 16 + l16) * Dmod;
  const unsigned int lds_off = (unsigned int)(__SIZE_TYPE__)wtile;

  for (int kp = 0; kp < Dmod; kp += KPAN) {
    const _Float16* wsrc = W16 + (size_t)colbase * Dmod + kp;
#if HAVE_TDM
    if (wave == 0) tdm_load_panel(lds_off, wsrc);
#else
    for (int c = threadIdx.x; c < 64 * (KPAN / 8); c += 256) {
      const int n = c >> 5, kk = (c & 31) * 8;
      *(v8h*)&wtile[n * WROW + kk] = *(const v8h*)(wsrc + (size_t)n * Dmod + kk);
    }
#endif
    __syncthreads();
#pragma unroll 2
    for (int k0 = 0; k0 < KPAN; k0 += 32) {
      v16h a0 = load_a_f16(arow0 + kp + k0, hh);
      v16h a1 = load_a_f16(arow1 + kp + k0, hh);
#pragma unroll
      for (int s = 0; s < 4; ++s) {
        // B-fragment: col = s*16+l16, elem e<8: K = k0 + hh*8 + e, e>=8: K = k0 + 16 + hh*8 + (e-8)
        const _Float16* lp = &wtile[(s * 16 + l16) * WROW + k0 + hh * 8];
        v16h b = join8(*(const v8h*)lp, *(const v8h*)(lp + 16));
        acc.a[0][s] = wmma16(a0, b, acc.a[0][s]);
        acc.a[1][s] = wmma16(a1, b, acc.a[1][s]);
      }
    }
    __syncthreads();
  }
}

// ---------------------------------------------------------------------------
// Kernel 1: q = x@Wq^T, k = x@Wk^T, v = x@Wv^T  (blockIdx.z selects weight)
// q,k -> (B,H,N,hd) f16 ; v -> (B,H,hd,N) f16
// ---------------------------------------------------------------------------
__global__ __launch_bounds__(256) void qkv_kernel(
    const _Float16* __restrict__ xh, const _Float16* __restrict__ wq,
    const _Float16* __restrict__ wk, const _Float16* __restrict__ wv,
    _Float16* __restrict__ qh, _Float16* __restrict__ kh,
    _Float16* __restrict__ vt) {
  __shared__ _Float16 wtile[64 * WROW];
  const int lane = threadIdx.x & 31;
  const int wave = threadIdx.x >> 5;
  const int hh = lane >> 4, l16 = lane & 15;
  const int which = blockIdx.z;
  const _Float16* W = (which == 0) ? wq : (which == 1 ? wk : wv);
  const int rowbase = (blockIdx.x * 8 + wave) * 32;
  const int colbase = blockIdx.y * 64;

  GemmAcc acc;
  gemm_core(xh, W, wtile, rowbase, colbase, wave, lane, acc);

#pragma unroll
  for (int g = 0; g < 2; ++g)
#pragma unroll
    for (int s = 0; s < 4; ++s) {
      const int cg = colbase + s * 16 + l16;
      const int h = cg >> 6, d = cg & 63;
#pragma unroll
      for (int r = 0; r < 8; ++r) {
        const int m = rowbase + g * 16 + r + 8 * hh;
        const int bb = m >> 9, n = m & 511;
        const _Float16 val = (_Float16)acc.a[g][s][r];
        if (which == 2)
          vt[(((size_t)bb * Hn + h) * HD + d) * Nseq + n] = val;  // V^T
        else if (which == 0)
          qh[(((size_t)bb * Hn + h) * Nseq + n) * HD + d] = val;
        else
          kh[(((size_t)bb * Hn + h) * Nseq + n) * HD + d] = val;
      }
    }
}

// ---------------------------------------------------------------------------
// Kernel 2: flash attention per (b,h); each wave owns 16 queries.
// S^T orientation => softmax stats need only one __shfl_xor(16).
// ---------------------------------------------------------------------------
__global__ __launch_bounds__(256) void attn_kernel(
    const _Float16* __restrict__ qh, const _Float16* __restrict__ kh,
    const _Float16* __restrict__ vt, const float* __restrict__ bs,
    _Float16* __restrict__ of) {
  __shared__ _Float16 lds_p[8][16][40];

  const int lane = threadIdx.x & 31;
  const int wave = threadIdx.x >> 5;
  const int hh = lane >> 4, l16 = lane & 15;
  const int bh = blockIdx.x;
  const int b = bh >> 4, h = bh & 15;
  const int qbase = (blockIdx.y * 8 + wave) * 16;
  const int q = qbase + l16;

  const _Float16* qrow = qh + ((size_t)bh * Nseq + q) * HD;
  v16h bq0 = load_a_f16(qrow, hh);
  v16h bq1 = load_a_f16(qrow + 32, hh);

  v8f acc[4];
#pragma unroll
  for (int t = 0; t < 4; ++t)
#pragma unroll
    for (int e = 0; e < 8; ++e) acc[t][e] = 0.0f;

  float m_i = -1e30f, l_i = 0.0f;
  const float* bias_row = bs + ((size_t)b * Nseq + q) * Nseq;

  for (int j0 = 0; j0 < Nseq; j0 += 32) {
    float s[2][8];
#pragma unroll
    for (int c = 0; c < 2; ++c) {
      const int key = j0 + c * 16 + l16;
      const _Float16* krow = kh + ((size_t)bh * Nseq + key) * HD;
      v8f st;
#pragma unroll
      for (int e = 0; e < 8; ++e) st[e] = 0.0f;
      st = wmma16(load_a_f16(krow, hh), bq0, st);
      st = wmma16(load_a_f16(krow + 32, hh), bq1, st);
      const int kb = j0 + c * 16 + hh * 8;
      v4f b0 = *(const v4f*)(bias_row + kb);
      v4f b1 = *(const v4f*)(bias_row + kb + 4);
#pragma unroll
      for (int r = 0; r < 4; ++r) {
        s[c][r]     = st[r] * SCALE + b0[r];
        s[c][r + 4] = st[r + 4] * SCALE + b1[r];
      }
    }

    float tm = s[0][0];
#pragma unroll
    for (int c = 0; c < 2; ++c)
#pragma unroll
      for (int r = 0; r < 8; ++r) tm = fmaxf(tm, s[c][r]);
    tm = fmaxf(tm, __shfl_xor(tm, 16, 32));
    const float m_new = fmaxf(m_i, tm);
    const float alpha = __expf(m_i - m_new);
    l_i *= alpha;
#pragma unroll
    for (int t = 0; t < 4; ++t)
#pragma unroll
      for (int e = 0; e < 8; ++e) acc[t][e] *= alpha;
#pragma unroll
    for (int c = 0; c < 2; ++c)
#pragma unroll
      for (int r = 0; r < 8; ++r) {
        const float p = __expf(s[c][r] - m_new);
        l_i += p;
        lds_p[wave][l16][c * 16 + hh * 8 + r] = (_Float16)p;
      }
    m_i = m_new;

    v8h p0 = *(const v8h*)(&lds_p[wave][l16][hh * 8]);
    v8h p1 = *(const v8h*)(&lds_p[wave][l16][16 + hh * 8]);
    const v16h bp = join8(p0, p1);

#pragma unroll
    for (int t = 0; t < 4; ++t) {
      const int d = t * 16 + l16;
      const _Float16* vrow = vt + ((size_t)bh * HD + d) * Nseq + j0;
      acc[t] = wmma16(load_a_f16(vrow, hh), bp, acc[t]);
    }
  }

  float l_tot = l_i + __shfl_xor(l_i, 16, 32);
  const float inv = (l_tot > 0.0f) ? (1.0f / l_tot) : 0.0f;

  _Float16* orow = of + ((size_t)b * Nseq + q) * Dmod + h * HD;
#pragma unroll
  for (int t = 0; t < 4; ++t)
#pragma unroll
    for (int r = 0; r < 8; ++r)
      orow[t * 16 + r + 8 * hh] = (_Float16)(acc[t][r] * inv);
}

// ---------------------------------------------------------------------------
// Kernel 3: out = o_flat @ Wo^T   (f32 out)
// ---------------------------------------------------------------------------
__global__ __launch_bounds__(256) void oproj_kernel(
    const _Float16* __restrict__ of, const _Float16* __restrict__ wo,
    float* __restrict__ out) {
  __shared__ _Float16 wtile[64 * WROW];
  const int lane = threadIdx.x & 31;
  const int wave = threadIdx.x >> 5;
  const int hh = lane >> 4, l16 = lane & 15;
  const int rowbase = (blockIdx.x * 8 + wave) * 32;
  const int colbase = blockIdx.y * 64;

  GemmAcc acc;
  gemm_core(of, wo, wtile, rowbase, colbase, wave, lane, acc);

#pragma unroll
  for (int g = 0; g < 2; ++g)
#pragma unroll
    for (int s = 0; s < 4; ++s) {
      const int c = colbase + s * 16 + l16;
#pragma unroll
      for (int r = 0; r < 8; ++r) {
        const int m = rowbase + g * 16 + r + 8 * hh;
        out[(size_t)m * Dmod + c] = acc.a[g][s][r];
      }
    }
}

// ---------------------------------------------------------------------------
extern "C" void kernel_launch(void* const* d_in, const int* in_sizes, int n_in,
                              void* d_out, int out_size, void* d_ws,
                              size_t ws_size, hipStream_t stream) {
  const float* x  = (const float*)d_in[0];
  const float* sp = (const float*)d_in[1];
  const float* ed = (const float*)d_in[2];
  const float* Wq = (const float*)d_in[3];
  const float* Wk = (const float*)d_in[4];
  const float* Wv = (const float*)d_in[5];
  const float* Wo = (const float*)d_in[6];
  float* out = (float*)d_out;

  const size_t xelems = (size_t)BN * Dmod;     // 8M
  const size_t welems = (size_t)Dmod * Dmod;   // 1M
  const size_t belems = (size_t)Bdim * Nseq * Nseq;  // 4M

  _Float16* xh  = (_Float16*)d_ws;
  _Float16* wq16 = xh + xelems;
  _Float16* wk16 = wq16 + welems;
  _Float16* wv16 = wk16 + welems;
  _Float16* wo16 = wv16 + welems;
  _Float16* qh  = wo16 + welems;
  _Float16* kh  = qh + xelems;
  _Float16* vt  = kh + xelems;
  _Float16* of  = vt + xelems;
  float*    bsum = (float*)(of + xelems);

  cvt16_kernel<<<2048, 256, 0, stream>>>(x, xh, (int)(xelems / 4));
  cvt16_kernel<<<512, 256, 0, stream>>>(Wq, wq16, (int)(welems / 4));
  cvt16_kernel<<<512, 256, 0, stream>>>(Wk, wk16, (int)(welems / 4));
  cvt16_kernel<<<512, 256, 0, stream>>>(Wv, wv16, (int)(welems / 4));
  cvt16_kernel<<<512, 256, 0, stream>>>(Wo, wo16, (int)(welems / 4));
  biassum_kernel<<<2048, 256, 0, stream>>>(sp, ed, bsum, (int)(belems / 4));

  qkv_kernel<<<dim3(BN / 256, Dmod / 64, 3), 256, 0, stream>>>(
      xh, wq16, wk16, wv16, qh, kh, vt);
  attn_kernel<<<dim3(Bdim * Hn, Nseq / 128), 256, 0, stream>>>(qh, kh, vt,
                                                               bsum, of);
  oproj_kernel<<<dim3(BN / 256, Dmod / 64), 256, 0, stream>>>(of, wo16, out);
}
